// LSTM_60842506715793
// MI455X (gfx1250) — matrix operand order repaired and checked by reading, not compiled, not run on hardware
//
#include <hip/hip_runtime.h>
#include <hip/hip_bf16.h>
#include <cstdint>

// Problem dims (fixed by the reference)
#define BATCH 1024
#define IN_D  1024
#define HID_D 4096
#define OUT_D 4096
#define XK    (IN_D + HID_D)     // 5120
#define GN    (4 * HID_D)        // 16384

// GEMM tiling: WG tile 128x128, KT=64. 8 waves in 4x2; wave tile 32x64.
#define BM 128
#define BN 128
#define KT 64
#define LDS_STRIDE 72            // KT + 8 halves padding (144B rows, 16B-aligned)

typedef __attribute__((ext_vector_type(16))) __bf16 v16bf;
typedef __attribute__((ext_vector_type(8)))  float  v8f;
typedef int v4i __attribute__((vector_size(4 * sizeof(int))));

#if __has_builtin(__builtin_amdgcn_global_load_async_to_lds_b128)
#define HAVE_ASYNC_LDS 1
#else
#define HAVE_ASYNC_LDS 0
#endif

#define GBL_AS __attribute__((address_space(1)))
#define LDS_AS __attribute__((address_space(3)))

union FragU {
    uint4 q[2];
    v16bf v;
};

__device__ __forceinline__ unsigned short f2bf(float f) {
    unsigned u = __float_as_uint(f);
    unsigned r = u + 0x7FFFu + ((u >> 16) & 1u);   // round-to-nearest-even
    return (unsigned short)(r >> 16);
}

__device__ __forceinline__ float sigmoidf(float x) {
    return 1.0f / (1.0f + expf(-x));
}

// 16B global -> LDS copy. Async (ASYNCcnt, no data VGPRs) when available.
__device__ __forceinline__ void stage16(const unsigned short* g, unsigned short* l) {
#if HAVE_ASYNC_LDS
    __builtin_amdgcn_global_load_async_to_lds_b128(
        (GBL_AS v4i*)(unsigned long long)(uintptr_t)g,
        (LDS_AS v4i*)(unsigned int)(uintptr_t)l, 0, 0);
#else
    *(uint4*)l = *(const uint4*)g;
#endif
}

__device__ __forceinline__ void stage_wait() {
#if HAVE_ASYNC_LDS
#if __has_builtin(__builtin_amdgcn_s_wait_asynccnt)
    __builtin_amdgcn_s_wait_asynccnt(0);
#else
    asm volatile("s_wait_asynccnt 0" ::: "memory");
#endif
#endif
}

// ---------------------------------------------------------------------------
// fp32 -> bf16 conversion (grid-stride)
// ---------------------------------------------------------------------------
__global__ void cvt_f32_bf16_kernel(const float* __restrict__ src,
                                    unsigned short* __restrict__ dst, int n) {
    for (int i = blockIdx.x * blockDim.x + threadIdx.x; i < n;
         i += gridDim.x * blockDim.x) {
        dst[i] = f2bf(src[i]);
    }
}

// x = concat(i, h) packed to bf16: [BATCH][XK]
__global__ void pack_x_kernel(const float* __restrict__ in_i,
                              const float* __restrict__ in_h,
                              unsigned short* __restrict__ x_bf) {
    int idx = blockIdx.x * blockDim.x + threadIdx.x;
    int total = BATCH * XK;
    if (idx >= total) return;
    int b = idx / XK;
    int k = idx - b * XK;
    float v = (k < IN_D) ? in_i[(size_t)b * IN_D + k]
                         : in_h[(size_t)b * HID_D + (k - IN_D)];
    x_bf[idx] = f2bf(v);
}

// ---------------------------------------------------------------------------
// C[M][N] = A[M][K] * B[N][K]^T   (both operands bf16, f32 accumulate)
// BM=BN=128 per workgroup; 8 waves (4x2), each wave a 32x64 tile (2x4 WMMA)
// ---------------------------------------------------------------------------
__global__ __launch_bounds__(256) void gemm_bf16_nt(
    const unsigned short* __restrict__ A,
    const unsigned short* __restrict__ B,
    float* __restrict__ C,
    int M, int N, int K) {

    __shared__ __align__(16) unsigned short sA[BM * LDS_STRIDE];
    __shared__ __align__(16) unsigned short sB[BN * LDS_STRIDE];

    const int tid  = threadIdx.x;
    const int wave = tid >> 5;        // wave32
    const int lane = tid & 31;
    const int bm   = blockIdx.y * BM;
    const int bn   = blockIdx.x * BN;

    const int wm = (wave >> 1) * 32;  // 0,32,64,96
    const int wn = (wave & 1) * 64;   // 0,64

    const int lhalf = lane >> 4;      // 0 or 1
    const int lrow  = lane & 15;

    // staging coordinates: 1024 uint4 per tile, 4 per thread
    const int s_c8 = (tid & 7) * 8;   // half offset inside 64-half row

    v8f acc[2][4];
#pragma unroll
    for (int mi = 0; mi < 2; ++mi)
#pragma unroll
        for (int ni = 0; ni < 4; ++ni)
            acc[mi][ni] = (v8f)(0.0f);

    for (int k0 = 0; k0 < K; k0 += KT) {
        // ---- stage A tile: BM x KT halves (1024 uint4, 4 per thread) ----
#pragma unroll
        for (int i = 0; i < 4; ++i) {
            int r = (tid >> 3) + i * 32;
            stage16(A + (size_t)(bm + r) * K + k0 + s_c8,
                    sA + r * LDS_STRIDE + s_c8);
        }
        // ---- stage B tile: BN x KT halves (1024 uint4, 4 per thread) ----
#pragma unroll
        for (int i = 0; i < 4; ++i) {
            int r = (tid >> 3) + i * 32;
            stage16(B + (size_t)(bn + r) * K + k0 + s_c8,
                    sB + r * LDS_STRIDE + s_c8);
        }
        // prefetch next K-tile of the (L2-resident) B operand
        if (k0 + KT < K) {
            __builtin_prefetch(B + (size_t)(bn + (tid >> 3)) * K + k0 + KT + s_c8, 0, 1);
        }
        stage_wait();
        __syncthreads();

#pragma unroll
        for (int kk = 0; kk < KT; kk += 32) {
            // A fragment (16x32 bf16): lanes 0-15 -> K {0..7,16..23},
            //                          lanes 16-31 -> K {8..15,24..31}
            v16bf afrag[2];
#pragma unroll
            for (int mi = 0; mi < 2; ++mi) {
                const unsigned short* p =
                    sA + (wm + mi * 16 + lrow) * LDS_STRIDE + kk + lhalf * 8;
                FragU fu;
                fu.q[0] = *(const uint4*)p;
                fu.q[1] = *(const uint4*)(p + 16);
                afrag[mi] = fu.v;
            }
            // B fragment (32x16 bf16): lane n<16 -> K {0..7,16..23} of col n,
            //                          lane n+16 -> K {8..15,24..31} of col n
            v16bf bfrag[4];
#pragma unroll
            for (int ni = 0; ni < 4; ++ni) {
                const unsigned short* p =
                    sB + (wn + ni * 16 + lrow) * LDS_STRIDE + kk + lhalf * 8;
                FragU fu;
                fu.q[0] = *(const uint4*)p;
                fu.q[1] = *(const uint4*)(p + 16);
                bfrag[ni] = fu.v;
            }
#pragma unroll
            for (int mi = 0; mi < 2; ++mi)
#pragma unroll
                for (int ni = 0; ni < 4; ++ni)
                    acc[mi][ni] = __builtin_amdgcn_wmma_f32_16x16x32_bf16(
                        false, afrag[mi], false, bfrag[ni],
                        (short)0, acc[mi][ni], false, false);
        }
        __syncthreads();
    }

    // D layout: VGPR r -> row = r + 8*lhalf, col = lane & 15
#pragma unroll
    for (int mi = 0; mi < 2; ++mi)
#pragma unroll
        for (int ni = 0; ni < 4; ++ni) {
            int col = bn + wn + ni * 16 + lrow;
#pragma unroll
            for (int r = 0; r < 8; ++r) {
                int row = bm + wm + mi * 16 + r + 8 * lhalf;
                C[(size_t)row * N + col] = acc[mi][ni][r];
            }
        }
}

// ---------------------------------------------------------------------------
// LSTM pointwise: biases + activations, emits it/ht/ct (fp32) and ht (bf16)
// ---------------------------------------------------------------------------
__global__ void lstm_pointwise_kernel(const float* __restrict__ gates,
                                      const float* __restrict__ c_in,
                                      const float* __restrict__ b_f,
                                      const float* __restrict__ b_i,
                                      const float* __restrict__ b_c,
                                      const float* __restrict__ b_o,
                                      float* __restrict__ out_it,
                                      float* __restrict__ out_ht,
                                      float* __restrict__ out_ct,
                                      unsigned short* __restrict__ ht_bf) {
    int idx = blockIdx.x * blockDim.x + threadIdx.x;
    if (idx >= BATCH * HID_D) return;
    int b = idx >> 12;            // / 4096
    int j = idx & (HID_D - 1);

    const float* g = gates + (size_t)b * GN;
    float f_pre = g[j]              + b_f[j];
    float i_pre = g[HID_D + j]      + b_i[j];
    float c_pre = g[2 * HID_D + j]  + b_c[j];
    float o_pre = g[3 * HID_D + j]  + b_o[j];

    float ft    = sigmoidf(f_pre);
    float it    = sigmoidf(i_pre);
    float ctbar = tanhf(c_pre);
    float ct    = ft * c_in[idx] + it * ctbar;
    float ot    = sigmoidf(o_pre);
    float ht    = ot * tanhf(ct);

    out_it[idx] = it;
    out_ht[idx] = ht;
    out_ct[idx] = ct;
    ht_bf[idx]  = f2bf(ht);
}

// ---------------------------------------------------------------------------
// Row-wise log_softmax over OUT_D=4096 (adds bias); one block per row
// ---------------------------------------------------------------------------
__global__ __launch_bounds__(256) void log_softmax_kernel(
    const float* __restrict__ logits, const float* __restrict__ bias,
    float* __restrict__ out) {
    __shared__ float red[256];
    const int row = blockIdx.x;
    const int tid = threadIdx.x;
    const float* lr = logits + (size_t)row * OUT_D;

    float mx = -3.402823466e+38f;
    for (int j = tid; j < OUT_D; j += 256)
        mx = fmaxf(mx, lr[j] + bias[j]);
    red[tid] = mx;
    __syncthreads();
    for (int s = 128; s > 0; s >>= 1) {
        if (tid < s) red[tid] = fmaxf(red[tid], red[tid + s]);
        __syncthreads();
    }
    mx = red[0];
    __syncthreads();

    float sum = 0.0f;
    for (int j = tid; j < OUT_D; j += 256)
        sum += expf(lr[j] + bias[j] - mx);
    red[tid] = sum;
    __syncthreads();
    for (int s = 128; s > 0; s >>= 1) {
        if (tid < s) red[tid] += red[tid + s];
        __syncthreads();
    }
    float lse = mx + logf(red[0]);

    float* orow = out + (size_t)row * OUT_D;
    for (int j = tid; j < OUT_D; j += 256)
        orow[j] = lr[j] + bias[j] - lse;
}

// ---------------------------------------------------------------------------
extern "C" void kernel_launch(void* const* d_in, const int* in_sizes, int n_in,
                              void* d_out, int out_size, void* d_ws, size_t ws_size,
                              hipStream_t stream) {
    (void)in_sizes; (void)n_in; (void)out_size; (void)ws_size;

    const float* in_i  = (const float*)d_in[0];
    const float* in_h  = (const float*)d_in[1];
    const float* in_c  = (const float*)d_in[2];
    const float* Wf    = (const float*)d_in[3];
    const float* bf_   = (const float*)d_in[4];
    const float* Wi    = (const float*)d_in[5];
    const float* bi_   = (const float*)d_in[6];
    const float* Wc    = (const float*)d_in[7];
    const float* bc_   = (const float*)d_in[8];
    const float* Wo    = (const float*)d_in[9];
    const float* bo_   = (const float*)d_in[10];
    const float* Wout  = (const float*)d_in[11];
    const float* bout  = (const float*)d_in[12];

    float* out = (float*)d_out;
    float* out_it  = out;
    float* out_ht  = out + (size_t)BATCH * HID_D;
    float* out_ct  = out + 2 * (size_t)BATCH * HID_D;
    float* out_lsm = out + 3 * (size_t)BATCH * HID_D;

    // Workspace carve-up (bytes)
    char* ws = (char*)d_ws;
    unsigned short* W_bf    = (unsigned short*)(ws);                       // 16384*5120*2
    unsigned short* Wout_bf = (unsigned short*)(ws + 167772160ull);        // 4096*4096*2
    unsigned short* x_bf    = (unsigned short*)(ws + 201326592ull);        // 1024*5120*2
    unsigned short* ht_bf   = (unsigned short*)(ws + 211812352ull);        // 1024*4096*2
    float*          gates   = (float*)(ws + 220200960ull);                 // 1024*16384*4
    float*          logits  = (float*)(ws + 287309824ull);                 // 1024*4096*4

    const int threads = 256;
    const int cvt_blocks = 2048;

    // 1) pack x = [i | h] as bf16
    {
        int total = BATCH * XK;
        pack_x_kernel<<<(total + threads - 1) / threads, threads, 0, stream>>>(
            in_i, in_h, x_bf);
    }
    // 2) weights -> bf16 (W_all stacked [Wf;Wi;Wc;Wo], 168MB fits in 192MB L2)
    {
        const int wsz = HID_D * XK;  // 20,971,520 each
        cvt_f32_bf16_kernel<<<cvt_blocks, threads, 0, stream>>>(Wf, W_bf + 0ull * wsz, wsz);
        cvt_f32_bf16_kernel<<<cvt_blocks, threads, 0, stream>>>(Wi, W_bf + 1ull * wsz, wsz);
        cvt_f32_bf16_kernel<<<cvt_blocks, threads, 0, stream>>>(Wc, W_bf + 2ull * wsz, wsz);
        cvt_f32_bf16_kernel<<<cvt_blocks, threads, 0, stream>>>(Wo, W_bf + 3ull * wsz, wsz);
        cvt_f32_bf16_kernel<<<cvt_blocks, threads, 0, stream>>>(Wout, Wout_bf, OUT_D * HID_D);
    }
    // 3) gates = x @ W_all^T   [1024 x 16384], K = 5120
    {
        dim3 grid(GN / BN, BATCH / BM);
        gemm_bf16_nt<<<grid, threads, 0, stream>>>(x_bf, W_bf, gates,
                                                   BATCH, GN, XK);
    }
    // 4) pointwise gates -> it/ht/ct (+ ht as bf16)
    {
        int total = BATCH * HID_D;
        lstm_pointwise_kernel<<<(total + threads - 1) / threads, threads, 0, stream>>>(
            gates, in_c, bf_, bi_, bc_, bo_, out_it, out_ht, out_ct, ht_bf);
    }
    // 5) logits = ht @ Wout^T   [1024 x 4096], K = 4096
    {
        dim3 grid(OUT_D / BN, BATCH / BM);
        gemm_bf16_nt<<<grid, threads, 0, stream>>>(ht_bf, Wout_bf, logits,
                                                   BATCH, OUT_D, HID_D);
    }
    // 6) log_softmax rows (adds bout)
    log_softmax_kernel<<<BATCH, threads, 0, stream>>>(logits, bout, out_lsm);
}
